// GlobalModel_15513421873649
// MI455X (gfx1250) — matrix operand order repaired and checked by reading, not compiled, not run on hardware
//
#include <hip/hip_runtime.h>
#include <hip/hip_bf16.h>
#include <stddef.h>

// ---------------------------------------------------------------------------
// GlobalModel block for MI455X (gfx1250), wave32 + WMMA bf16.
// ---------------------------------------------------------------------------

typedef __attribute__((ext_vector_type(16))) __bf16 v16bf;
typedef __attribute__((ext_vector_type(8)))  float  v8f;

#define N_NODES   500000
#define N_GRAPHS  4096
#define LN_EPS    1e-5f

// LDS layout. bf16 rows padded to 136 halfs (272B = 17*16B) to stagger banks
// while keeping every 8-half run 16B aligned for b128 LDS loads.
#define WSTRIDE 136
#define SM_WT1  0
#define SM_WT2  (SM_WT1 + 128 * WSTRIDE * 2)
#define SM_WT3  (SM_WT2 + 128 * WSTRIDE * 2)
#define SM_HA   (SM_WT3 + 64 * WSTRIDE * 2)
#define SM_HB   (SM_HA + 16 * WSTRIDE * 2)
#define SM_HC   (SM_HB + 16 * WSTRIDE * 2)   // f32 16 x 68
#define SM_SG   (SM_HC + 16 * 68 * 4)
#define SMEM_BYTES (SM_SG + 64)

// workspace layout (bytes)
#define WS_AGG  0
#define WS_WT1  (WS_AGG + N_GRAPHS * 64 * 4)          // 1,048,576
#define WS_WT2  (WS_WT1 + 128 * 128 * 2)
#define WS_WT3  (WS_WT2 + 128 * 128 * 2)
#define WS_WT4  (WS_WT3 + 64 * 128 * 2)
#define WS_WT5  (WS_WT4 + 128 * 128 * 2)
#define WS_WT6  (WS_WT5 + 128 * 128 * 2)

__device__ __forceinline__ unsigned short f2bf(float f) {
  unsigned int x = __float_as_uint(f);
  unsigned int r = x + 0x7fffu + ((x >> 16) & 1u);   // round-to-nearest-even
  return (unsigned short)(r >> 16);
}

union ABReg { uint4 q[2]; v16bf v; };
union Pack8 { unsigned short s[8]; uint4 q; };

// One 16-row tile MLP layer: hin[16][128] bf16 (LDS) x wt[ncols][128] bf16
// (LDS, transposed weights) + bias -> out. Each wave owns one 16-col tile.
// RELU / OUTF32 are compile-time so the epilogue is straight-line code.
template <bool RELU, bool OUTF32>
__device__ __forceinline__ void mlp_layer(const unsigned short* hin,
                                          const unsigned short* wt,
                                          float bN,            // bias[col], preloaded
                                          unsigned short* hout_bf,
                                          float* hout_f,
                                          int ncols, int tid) {
  const int lane = tid & 31;
  const int wv   = tid >> 5;
  if (wv * 16 < ncols) {                 // wave-uniform: EXEC stays all-ones
    const int n   = lane & 15;
    const int hi  = lane >> 4;
    const int col = wv * 16 + n;
    v8f c;
#pragma unroll
    for (int j = 0; j < 8; ++j) c[j] = bN;       // C layout: lane holds fixed N
    const unsigned short* arow = hin + n * WSTRIDE;    // A: M = lane&15
    const unsigned short* brow = wt + col * WSTRIDE;   // B: N = lane&15
#pragma unroll
    for (int kk = 0; kk < 4; ++kk) {             // K = 128 = 4 x 32
      ABReg a, b;
      const int aoff = kk * 32 + hi * 8;         // A runs: [aoff,+8) and [aoff+16,+8)
      a.q[0] = *(const uint4*)(arow + aoff);
      a.q[1] = *(const uint4*)(arow + aoff + 16);
      const int boff = kk * 32 + hi * 8;         // B runs: [boff,+8) and [boff+16,+8)
      b.q[0] = *(const uint4*)(brow + boff);
      b.q[1] = *(const uint4*)(brow + boff + 16);
      c = __builtin_amdgcn_wmma_f32_16x16x32_bf16(false, a.v, false, b.v,
                                                  (short)0, c, false, false);
    }
#pragma unroll
    for (int j = 0; j < 8; ++j) {
      float v = c[j];
      if (RELU) v = fmaxf(v, 0.0f);
      const int m = hi * 8 + j;                  // D rows: 8*hi + j
      if (OUTF32) hout_f[m * 68 + col] = v;
      else        hout_bf[m * WSTRIDE + col] = f2bf(v);
    }
  }
}

// Copy transposed bf16 weights (dense [row][128]) from global/L2 into LDS.
__device__ __forceinline__ void load_weights(unsigned char* smem,
                                             const unsigned short* __restrict__ g1,
                                             const unsigned short* __restrict__ g2,
                                             const unsigned short* __restrict__ g3,
                                             int tid) {
  unsigned short* d1 = (unsigned short*)(smem + SM_WT1);
  unsigned short* d2 = (unsigned short*)(smem + SM_WT2);
  unsigned short* d3 = (unsigned short*)(smem + SM_WT3);
  for (int idx = tid; idx < 320 * 16; idx += 256) {   // 128+128+64 rows, 16B chunks
    const int row = idx >> 4, q = idx & 15;
    const unsigned short* src;
    unsigned short* dst;
    if (row < 128)      { src = g1 + row * 128;         dst = d1 + row * WSTRIDE; }
    else if (row < 256) { src = g2 + (row - 128) * 128; dst = d2 + (row - 128) * WSTRIDE; }
    else                { src = g3 + (row - 256) * 128; dst = d3 + (row - 256) * WSTRIDE; }
    *(uint4*)(dst + q * 8) = *(const uint4*)(src + q * 8);
  }
}

// --- Kernel 0: zero agg + convert+transpose six weight matrices to bf16 -----
__device__ __forceinline__ void conv_wt(const float* __restrict__ w,
                                        unsigned short* __restrict__ wt,
                                        int OUT, int gid, int stride) {
  for (int i = gid; i < OUT * 128; i += stride) {
    const int o = i >> 7, k = i & 127;
    wt[o * 128 + k] = f2bf(w[k * OUT + o]);
  }
}

__global__ void prep_kernel(const float* __restrict__ w1, const float* __restrict__ w2,
                            const float* __restrict__ w3, const float* __restrict__ w4,
                            const float* __restrict__ w5, const float* __restrict__ w6,
                            unsigned char* __restrict__ ws) {
  const int gid = blockIdx.x * blockDim.x + threadIdx.x;
  const int stride = gridDim.x * blockDim.x;
  float* agg = (float*)(ws + WS_AGG);
  for (int i = gid; i < N_GRAPHS * 64; i += stride) agg[i] = 0.0f;
  conv_wt(w1, (unsigned short*)(ws + WS_WT1), 128, gid, stride);
  conv_wt(w2, (unsigned short*)(ws + WS_WT2), 128, gid, stride);
  conv_wt(w3, (unsigned short*)(ws + WS_WT3),  64, gid, stride);
  conv_wt(w4, (unsigned short*)(ws + WS_WT4), 128, gid, stride);
  conv_wt(w5, (unsigned short*)(ws + WS_WT5), 128, gid, stride);
  conv_wt(w6, (unsigned short*)(ws + WS_WT6),  64, gid, stride);
}

// --- Kernel 1: node MLP + LayerNorm + segment-sum scatter -------------------
__global__ void __launch_bounds__(256)
node_phase_kernel(const float* __restrict__ x, const float* __restrict__ u,
                  const int* __restrict__ batch,
                  const unsigned short* __restrict__ wt1,
                  const unsigned short* __restrict__ wt2,
                  const unsigned short* __restrict__ wt3,
                  const float* __restrict__ b1, const float* __restrict__ b2,
                  const float* __restrict__ b3,
                  const float* __restrict__ ln_g, const float* __restrict__ ln_b,
                  float* __restrict__ agg) {
  extern __shared__ unsigned char smem[];
  const int tid = threadIdx.x;
  load_weights(smem, wt1, wt2, wt3, tid);

  unsigned short* HA = (unsigned short*)(smem + SM_HA);
  unsigned short* HB = (unsigned short*)(smem + SM_HB);
  float* HC = (float*)(smem + SM_HC);
  int* SG = (int*)(smem + SM_SG);

  const int ntiles = N_NODES / 16;     // 31250, exact
  const int row = tid >> 4;
  const int c8  = (tid & 15) * 8;
  const int l   = tid & 15;
  const int wv  = tid >> 5;
  const int bcol = wv * 16 + (tid & 15);

  // tile-invariant parameters -> registers (off the hot loop)
  const float bn1 = b1[bcol];
  const float bn2 = b2[bcol];
  const float bn3 = (wv < 4) ? b3[bcol] : 0.0f;
  float lg[4], lb[4];
#pragma unroll
  for (int k = 0; k < 4; ++k) { lg[k] = ln_g[l + 16 * k]; lb[k] = ln_b[l + 16 * k]; }
  __syncthreads();

  for (int tile = blockIdx.x; tile < ntiles; tile += gridDim.x) {
    if (tid < 16) SG[tid] = batch[tile * 16 + tid];
    __syncthreads();

    // gather [x | u[batch]] tile -> bf16 LDS
    const int node = tile * 16 + row;
    const int g = SG[row];
    const float* src = (c8 < 64) ? (x + (size_t)node * 64 + c8)
                                 : (u + (size_t)g * 64 + (c8 - 64));
    const int pnext = tile + gridDim.x;
    if (pnext < ntiles && c8 < 64)
      __builtin_prefetch(x + (size_t)(pnext * 16 + row) * 64 + c8, 0, 1);
    float4 f0 = ((const float4*)src)[0];
    float4 f1 = ((const float4*)src)[1];
    Pack8 pk;
    pk.s[0] = f2bf(f0.x); pk.s[1] = f2bf(f0.y);
    pk.s[2] = f2bf(f0.z); pk.s[3] = f2bf(f0.w);
    pk.s[4] = f2bf(f1.x); pk.s[5] = f2bf(f1.y);
    pk.s[6] = f2bf(f1.z); pk.s[7] = f2bf(f1.w);
    *(uint4*)(HA + row * WSTRIDE + c8) = pk.q;
    __syncthreads();

    mlp_layer<true, false>(HA, (unsigned short*)(smem + SM_WT1), bn1, HB, nullptr, 128, tid);
    __syncthreads();
    mlp_layer<true, false>(HB, (unsigned short*)(smem + SM_WT2), bn2, HA, nullptr, 128, tid);
    __syncthreads();
    mlp_layer<false, true>(HA, (unsigned short*)(smem + SM_WT3), bn3, nullptr, HC, 64, tid);
    __syncthreads();

    // LayerNorm over 64 feats (16 lanes x 4 each) + f32 atomic scatter-add
    float v[4];
#pragma unroll
    for (int k = 0; k < 4; ++k) v[k] = HC[row * 68 + l + 16 * k];
    float s  = v[0] + v[1] + v[2] + v[3];
    float ss = v[0] * v[0] + v[1] * v[1] + v[2] * v[2] + v[3] * v[3];
#pragma unroll
    for (int off = 1; off < 16; off <<= 1) {
      s  += __shfl_xor(s, off, 16);
      ss += __shfl_xor(ss, off, 16);
    }
    const float mean = s * (1.0f / 64.0f);
    const float var  = ss * (1.0f / 64.0f) - mean * mean;
    const float rstd = rsqrtf(var + LN_EPS);
    float* dst = agg + (size_t)SG[row] * 64;
#pragma unroll
    for (int k = 0; k < 4; ++k) {
      const float y = (v[k] - mean) * rstd * lg[k] + lb[k];
      unsafeAtomicAdd(dst + l + 16 * k, y);   // global_atomic_add_f32, L2-resident
    }
    __syncthreads();
  }
}

// --- Kernel 2: graph MLP + LayerNorm + residual -----------------------------
__global__ void __launch_bounds__(256)
graph_phase_kernel(const float* __restrict__ agg, const float* __restrict__ u,
                   const unsigned short* __restrict__ wt4,
                   const unsigned short* __restrict__ wt5,
                   const unsigned short* __restrict__ wt6,
                   const float* __restrict__ b4, const float* __restrict__ b5,
                   const float* __restrict__ b6,
                   const float* __restrict__ ln_g, const float* __restrict__ ln_b,
                   float* __restrict__ out) {
  extern __shared__ unsigned char smem[];
  const int tid = threadIdx.x;
  load_weights(smem, wt4, wt5, wt6, tid);

  unsigned short* HA = (unsigned short*)(smem + SM_HA);
  unsigned short* HB = (unsigned short*)(smem + SM_HB);
  float* HC = (float*)(smem + SM_HC);

  const int tile = blockIdx.x;           // 256 tiles of 16 graphs
  const int row = tid >> 4;
  const int c8  = (tid & 15) * 8;
  const int l   = tid & 15;
  const int wv  = tid >> 5;
  const int bcol = wv * 16 + (tid & 15);
  const int gr  = tile * 16 + row;

  const float bn4 = b4[bcol];
  const float bn5 = b5[bcol];
  const float bn6 = (wv < 4) ? b6[bcol] : 0.0f;

  const float* src = (c8 < 64) ? (agg + (size_t)gr * 64 + c8)
                               : (u + (size_t)gr * 64 + (c8 - 64));
  float4 f0 = ((const float4*)src)[0];
  float4 f1 = ((const float4*)src)[1];
  Pack8 pk;
  pk.s[0] = f2bf(f0.x); pk.s[1] = f2bf(f0.y);
  pk.s[2] = f2bf(f0.z); pk.s[3] = f2bf(f0.w);
  pk.s[4] = f2bf(f1.x); pk.s[5] = f2bf(f1.y);
  pk.s[6] = f2bf(f1.z); pk.s[7] = f2bf(f1.w);
  *(uint4*)(HA + row * WSTRIDE + c8) = pk.q;
  __syncthreads();

  mlp_layer<true, false>(HA, (unsigned short*)(smem + SM_WT1), bn4, HB, nullptr, 128, tid);
  __syncthreads();
  mlp_layer<true, false>(HB, (unsigned short*)(smem + SM_WT2), bn5, HA, nullptr, 128, tid);
  __syncthreads();
  mlp_layer<false, true>(HA, (unsigned short*)(smem + SM_WT3), bn6, nullptr, HC, 64, tid);
  __syncthreads();

  float v[4];
#pragma unroll
  for (int k = 0; k < 4; ++k) v[k] = HC[row * 68 + l + 16 * k];
  float s  = v[0] + v[1] + v[2] + v[3];
  float ss = v[0] * v[0] + v[1] * v[1] + v[2] * v[2] + v[3] * v[3];
#pragma unroll
  for (int off = 1; off < 16; off <<= 1) {
    s  += __shfl_xor(s, off, 16);
    ss += __shfl_xor(ss, off, 16);
  }
  const float mean = s * (1.0f / 64.0f);
  const float var  = ss * (1.0f / 64.0f) - mean * mean;
  const float rstd = rsqrtf(var + LN_EPS);
#pragma unroll
  for (int k = 0; k < 4; ++k) {
    const int col = l + 16 * k;
    const float y = (v[k] - mean) * rstd * ln_g[col] + ln_b[col];
    out[(size_t)gr * 64 + col] = y + u[(size_t)gr * 64 + col];
  }
}

// ---------------------------------------------------------------------------
extern "C" void kernel_launch(void* const* d_in, const int* in_sizes, int n_in,
                              void* d_out, int out_size, void* d_ws, size_t ws_size,
                              hipStream_t stream) {
  (void)in_sizes; (void)n_in; (void)out_size; (void)ws_size;
  const float* x     = (const float*)d_in[0];
  const float* u     = (const float*)d_in[1];
  const int*   batch = (const int*)d_in[2];
  const float* w1 = (const float*)d_in[3];  const float* b1 = (const float*)d_in[4];
  const float* w2 = (const float*)d_in[5];  const float* b2 = (const float*)d_in[6];
  const float* w3 = (const float*)d_in[7];  const float* b3 = (const float*)d_in[8];
  const float* ln1g = (const float*)d_in[9];  const float* ln1b = (const float*)d_in[10];
  const float* w4 = (const float*)d_in[11]; const float* b4 = (const float*)d_in[12];
  const float* w5 = (const float*)d_in[13]; const float* b5 = (const float*)d_in[14];
  const float* w6 = (const float*)d_in[15]; const float* b6 = (const float*)d_in[16];
  const float* ln2g = (const float*)d_in[17]; const float* ln2b = (const float*)d_in[18];

  unsigned char* ws = (unsigned char*)d_ws;
  float* agg = (float*)(ws + WS_AGG);
  unsigned short* wt1 = (unsigned short*)(ws + WS_WT1);
  unsigned short* wt2 = (unsigned short*)(ws + WS_WT2);
  unsigned short* wt3 = (unsigned short*)(ws + WS_WT3);
  unsigned short* wt4 = (unsigned short*)(ws + WS_WT4);
  unsigned short* wt5 = (unsigned short*)(ws + WS_WT5);
  unsigned short* wt6 = (unsigned short*)(ws + WS_WT6);

  prep_kernel<<<512, 256, 0, stream>>>(w1, w2, w3, w4, w5, w6, ws);
  node_phase_kernel<<<2048, 256, SMEM_BYTES, stream>>>(
      x, u, batch, wt1, wt2, wt3, b1, b2, b3, ln1g, ln1b, agg);
  graph_phase_kernel<<<256, 256, SMEM_BYTES, stream>>>(
      agg, u, wt4, wt5, wt6, b4, b5, b6, ln2g, ln2b, (float*)d_out);
}
